// AnimaMLP_7430293422694
// MI455X (gfx1250) — matrix operand order repaired and checked by reading, not compiled, not run on hardware
//
#include <hip/hip_runtime.h>
#include <math.h>

// ---------------- problem constants ----------------
#define MTOK   8192      // B*T
#define DDIM   2048
#define IDIM   1024
#define NEXP   8
#define NACT   5
#define NCAMPA 4
#define LPITCH 72        // LDS row pitch in bf16 elements (64 + 8 pad, 16B-aligned rows)

typedef __attribute__((ext_vector_type(16))) __bf16 v16bf;
typedef __attribute__((ext_vector_type(8)))  float  v8f;

// ---------------- helpers ----------------
__device__ inline unsigned short f2bf(float f) {
    union { float f; unsigned int u; } c; c.f = f;
    unsigned int u = c.u;
    unsigned int r = u + 0x7FFFu + ((u >> 16) & 1u);   // round-to-nearest-even
    return (unsigned short)(r >> 16);
}

__device__ inline float silu(float g) { return g / (1.0f + __expf(-g)); }

union FragBF { v16bf v; unsigned int d[8]; };

// A fragment (16x32 bf16) from row-major LDS tile [m][LPITCH].
// Lane l<16: M=l, K=0..15 (2 per VGPR); lane>=16: K=16..31.
__device__ inline v16bf load_a_frag(const unsigned short* lds, int mBase, int kOff, int lane) {
    const int m  = mBase + (lane & 15);
    const int kh = (lane >> 4) & 1;
    const unsigned short* rp = lds + m * LPITCH + kOff;
    FragBF f;
#pragma unroll
    for (int v = 0; v < 8; ++v) {
        int k = kh * 16 + 2 * v;
        f.d[v] = *(const unsigned int*)(rp + k);
    }
    return f.v;
}

// B fragment (32x16 bf16) from row-major B^T LDS tile [n][LPITCH] (k contiguous).
// Lane l<16: N=l, K=0..15 (2 per VGPR); lane>=16: K=16..31.
__device__ inline v16bf load_b_frag(const unsigned short* ldsT, int nBase, int kOff, int lane) {
    const unsigned short* rp = ldsT + (nBase + (lane & 15)) * LPITCH + kOff + ((lane >> 4) & 1) * 16;
    FragBF f;
#pragma unroll
    for (int v = 0; v < 8; ++v)
        f.d[v] = *(const unsigned int*)(rp + 2 * v);
    return f.v;
}

__device__ inline v8f wmma_bf16(v16bf a, v16bf b, v8f c) {
    return __builtin_amdgcn_wmma_f32_16x16x32_bf16(false, a, false, b, (short)0, c, false, false);
}

// ---------------- kernels ----------------
__global__ void zero_kernel(float* p) { *p = 0.0f; }

// straight FP32 -> BF16 (for x; stays row-major, it is always the A operand)
__global__ __launch_bounds__(256) void cvt_kernel(const float* __restrict__ s,
                                                  unsigned short* __restrict__ d, int n8) {
    int i = blockIdx.x * blockDim.x + threadIdx.x;
    if (i >= n8) return;
    const float4 a = ((const float4*)s)[(size_t)i * 2];
    const float4 b = ((const float4*)s)[(size_t)i * 2 + 1];
    union { uint4 q; unsigned short h[8]; } o;
    o.h[0] = f2bf(a.x); o.h[1] = f2bf(a.y); o.h[2] = f2bf(a.z); o.h[3] = f2bf(a.w);
    o.h[4] = f2bf(b.x); o.h[5] = f2bf(b.y); o.h[6] = f2bf(b.z); o.h[7] = f2bf(b.w);
    ((uint4*)d)[i] = o.q;
}

// Transposing FP32 -> BF16 convert: src (E,K,N) f32 -> dst (E,N,K) bf16.
__global__ __launch_bounds__(256) void cvtT_kernel(const float* __restrict__ src,
                                                   unsigned short* __restrict__ dst,
                                                   int K, int N) {
    const int e  = blockIdx.z;
    const int n0 = blockIdx.x * 32;
    const int k0 = blockIdx.y * 32;
    __shared__ unsigned short t[32][33];
    const int r = threadIdx.x >> 3, c4 = (threadIdx.x & 7) * 4;
    const float4 v = *(const float4*)(src + ((size_t)e * K + k0 + r) * N + n0 + c4);
    t[c4 + 0][r] = f2bf(v.x);
    t[c4 + 1][r] = f2bf(v.y);
    t[c4 + 2][r] = f2bf(v.z);
    t[c4 + 3][r] = f2bf(v.w);
    __syncthreads();
    union { uint2 q; unsigned short s[4]; } o;
    o.s[0] = t[r][c4 + 0]; o.s[1] = t[r][c4 + 1];
    o.s[2] = t[r][c4 + 2]; o.s[3] = t[r][c4 + 3];
    *(uint2*)(dst + ((size_t)e * N + n0 + r) * K + k0 + c4) = o.q;
}

// One block per token: scores = x . Wr ; softmax(/e); top-5 mask; renorm; fold sign.
__global__ __launch_bounds__(256) void router_kernel(const float* __restrict__ x,
                                                     const float* __restrict__ Wr,
                                                     float* __restrict__ wtok) {
    const int token = blockIdx.x;
    const float* xr = x + (size_t)token * DDIM;
    float acc[NEXP];
#pragma unroll
    for (int e = 0; e < NEXP; ++e) acc[e] = 0.0f;
    for (int d = threadIdx.x; d < DDIM; d += 256) {
        float xv = xr[d];
        const float* wr = Wr + (size_t)d * NEXP;
#pragma unroll
        for (int e = 0; e < NEXP; ++e) acc[e] += xv * wr[e];
    }
#pragma unroll
    for (int off = 16; off > 0; off >>= 1)
#pragma unroll
        for (int e = 0; e < NEXP; ++e) acc[e] += __shfl_down(acc[e], off, 32);

    __shared__ float rsm[8][NEXP];
    const int wave = threadIdx.x >> 5, lane = threadIdx.x & 31;
    if (lane == 0)
#pragma unroll
        for (int e = 0; e < NEXP; ++e) rsm[wave][e] = acc[e];
    __syncthreads();
    if (threadIdx.x == 0) {
        const float invT = 0.36787944117144233f; // 1/e
        float l[NEXP];
#pragma unroll
        for (int e = 0; e < NEXP; ++e) {
            float t = 0.0f;
            for (int w = 0; w < 8; ++w) t += rsm[w][e];
            l[e] = t * invT;
        }
        float mx = l[0];
        for (int e = 1; e < NEXP; ++e) mx = fmaxf(mx, l[e]);
        float p[NEXP], ps = 0.0f;
        for (int e = 0; e < NEXP; ++e) { p[e] = expf(l[e] - mx); ps += p[e]; }
        for (int e = 0; e < NEXP; ++e) p[e] /= ps;
        float w[NEXP], wsum = 0.0f;
        for (int e = 0; e < NEXP; ++e) {
            int rank = 0;
            for (int j = 0; j < NEXP; ++j)
                if (p[j] > p[e] || (p[j] == p[e] && j < e)) rank++;
            w[e] = (rank < NACT) ? p[e] : 0.0f;
            wsum += w[e];
        }
        float inv = 1.0f / (wsum + 1e-8f);
        for (int e = 0; e < NEXP; ++e)
            wtok[(size_t)token * NEXP + e] = w[e] * inv * ((e < NCAMPA) ? 1.0f : -1.0f);
    }
}

// GEMM1 fused: h = silu(x@Wg) * (x@Wu) per expert.
// Block tile 128x64, K-step 64; each wave owns a 32x32 quadrant (2m x 2n subtiles).
__global__ __launch_bounds__(256) void gemm1_kernel(const unsigned short* __restrict__ xb,
                                                    const unsigned short* __restrict__ wgT,
                                                    const unsigned short* __restrict__ wuT,
                                                    unsigned short* __restrict__ hb) {
    const int e  = blockIdx.z;
    const int n0 = blockIdx.x * 64;
    const int m0 = blockIdx.y * 128;
    const unsigned short* Bg = wgT + (size_t)e * IDIM * DDIM;
    const unsigned short* Bu = wuT + (size_t)e * IDIM * DDIM;

    __shared__ unsigned short lA[128 * LPITCH];
    __shared__ unsigned short lG[64 * LPITCH];
    __shared__ unsigned short lU[64 * LPITCH];

    const int tid  = threadIdx.x, lane = tid & 31, wave = tid >> 5;
    const int mGrp = wave & 3, nGrp = wave >> 2;       // 4 m-groups(32) x 2 n-groups(32)
    const int row  = tid >> 2, cB = (tid & 3) << 4;    // staging: row 0..63, col {0,16,32,48}

    v8f g[2][2] = {}, u[2][2] = {};

    for (int k0 = 0; k0 < DDIM; k0 += 64) {
        const unsigned short* aS0 = xb + (size_t)(m0 + row) * DDIM + k0 + cB;
        const unsigned short* aS1 = xb + (size_t)(m0 + 64 + row) * DDIM + k0 + cB;
        const unsigned short* gS  = Bg + (size_t)(n0 + row) * DDIM + k0 + cB;
        const unsigned short* uS  = Bu + (size_t)(n0 + row) * DDIM + k0 + cB;
        uint4 a00 = *(const uint4*)aS0, a01 = *(const uint4*)(aS0 + 8);
        uint4 a10 = *(const uint4*)aS1, a11 = *(const uint4*)(aS1 + 8);
        uint4 gg0 = *(const uint4*)gS,  gg1 = *(const uint4*)(gS + 8);
        uint4 uu0 = *(const uint4*)uS,  uu1 = *(const uint4*)(uS + 8);
        if (k0 + 64 < DDIM) {
            __builtin_prefetch(aS0 + 64, 0, 3);
            __builtin_prefetch(aS1 + 64, 0, 3);
            __builtin_prefetch(gS + 64, 0, 3);
            __builtin_prefetch(uS + 64, 0, 3);
        }
        *(uint4*)(lA + row * LPITCH + cB)            = a00;
        *(uint4*)(lA + row * LPITCH + cB + 8)        = a01;
        *(uint4*)(lA + (64 + row) * LPITCH + cB)     = a10;
        *(uint4*)(lA + (64 + row) * LPITCH + cB + 8) = a11;
        *(uint4*)(lG + row * LPITCH + cB)            = gg0;
        *(uint4*)(lG + row * LPITCH + cB + 8)        = gg1;
        *(uint4*)(lU + row * LPITCH + cB)            = uu0;
        *(uint4*)(lU + row * LPITCH + cB + 8)        = uu1;
        __syncthreads();

#pragma unroll
        for (int kk = 0; kk < 64; kk += 32) {
            v16bf a0  = load_a_frag(lA, mGrp * 32,      kk, lane);
            v16bf a1  = load_a_frag(lA, mGrp * 32 + 16, kk, lane);
            v16bf bg0 = load_b_frag(lG, nGrp * 32,      kk, lane);
            v16bf bg1 = load_b_frag(lG, nGrp * 32 + 16, kk, lane);
            v16bf bu0 = load_b_frag(lU, nGrp * 32,      kk, lane);
            v16bf bu1 = load_b_frag(lU, nGrp * 32 + 16, kk, lane);
            g[0][0] = wmma_bf16(a0, bg0, g[0][0]);
            g[0][1] = wmma_bf16(a0, bg1, g[0][1]);
            u[0][0] = wmma_bf16(a0, bu0, u[0][0]);
            u[0][1] = wmma_bf16(a0, bu1, u[0][1]);
            g[1][0] = wmma_bf16(a1, bg0, g[1][0]);
            g[1][1] = wmma_bf16(a1, bg1, g[1][1]);
            u[1][0] = wmma_bf16(a1, bu0, u[1][0]);
            u[1][1] = wmma_bf16(a1, bu1, u[1][1]);
        }
        __syncthreads();
    }

    const int rowOff = (lane >> 4) ? 8 : 0;
    const int col    = lane & 15;
#pragma unroll
    for (int mi = 0; mi < 2; ++mi) {
#pragma unroll
        for (int v = 0; v < 8; ++v) {
            int r = m0 + mGrp * 32 + mi * 16 + v + rowOff;
            size_t base = ((size_t)e * MTOK + r) * IDIM;
            hb[base + n0 + nGrp * 32 + col]      = f2bf(silu(g[mi][0][v]) * u[mi][0][v]);
            hb[base + n0 + nGrp * 32 + 16 + col] = f2bf(silu(g[mi][1][v]) * u[mi][1][v]);
        }
    }
}

// GEMM2: out = sum_e scale[m,e] * (h_e @ Wd_e); fused sum(out^2). wdT: [e][n=D][k=I].
// Block tile 128x64; wave owns 32x32 quadrant.
__global__ __launch_bounds__(256) void gemm2_kernel(const unsigned short* __restrict__ hb,
                                                    const unsigned short* __restrict__ wdT,
                                                    const float* __restrict__ wtok,
                                                    float* __restrict__ out,
                                                    float* __restrict__ sumsq) {
    const int n0 = blockIdx.x * 64;    // over D
    const int m0 = blockIdx.y * 128;   // over tokens

    __shared__ unsigned short lA[128 * LPITCH];
    __shared__ unsigned short lB[64 * LPITCH];
    __shared__ float lS[128 * NEXP];
    __shared__ float wred[8];

    const int tid  = threadIdx.x, lane = tid & 31, wave = tid >> 5;
    const int mGrp = wave & 3, nGrp = wave >> 2;
    const int row  = tid >> 2, cB = (tid & 3) << 4;

    for (int i = tid; i < 128 * NEXP; i += 256)
        lS[i] = wtok[(size_t)(m0 + (i >> 3)) * NEXP + (i & 7)];
    __syncthreads();

    v8f o[2][2] = {};
    const int rowOff = (lane >> 4) ? 8 : 0;
    const int col    = lane & 15;

    for (int e = 0; e < NEXP; ++e) {
        const unsigned short* A = hb  + (size_t)e * MTOK * IDIM;
        const unsigned short* B = wdT + (size_t)e * DDIM * IDIM;
        v8f t[2][2] = {};
        for (int k0 = 0; k0 < IDIM; k0 += 64) {
            const unsigned short* aS0 = A + (size_t)(m0 + row) * IDIM + k0 + cB;
            const unsigned short* aS1 = A + (size_t)(m0 + 64 + row) * IDIM + k0 + cB;
            const unsigned short* bS  = B + (size_t)(n0 + row) * IDIM + k0 + cB;
            uint4 a00 = *(const uint4*)aS0, a01 = *(const uint4*)(aS0 + 8);
            uint4 a10 = *(const uint4*)aS1, a11 = *(const uint4*)(aS1 + 8);
            uint4 b0  = *(const uint4*)bS,  b1  = *(const uint4*)(bS + 8);
            if (k0 + 64 < IDIM) {
                __builtin_prefetch(aS0 + 64, 0, 3);
                __builtin_prefetch(aS1 + 64, 0, 3);
                __builtin_prefetch(bS + 64, 0, 3);
            }
            *(uint4*)(lA + row * LPITCH + cB)            = a00;
            *(uint4*)(lA + row * LPITCH + cB + 8)        = a01;
            *(uint4*)(lA + (64 + row) * LPITCH + cB)     = a10;
            *(uint4*)(lA + (64 + row) * LPITCH + cB + 8) = a11;
            *(uint4*)(lB + row * LPITCH + cB)            = b0;
            *(uint4*)(lB + row * LPITCH + cB + 8)        = b1;
            __syncthreads();

#pragma unroll
            for (int kk = 0; kk < 64; kk += 32) {
                v16bf a0 = load_a_frag(lA, mGrp * 32,      kk, lane);
                v16bf a1 = load_a_frag(lA, mGrp * 32 + 16, kk, lane);
                v16bf bA = load_b_frag(lB, nGrp * 32,      kk, lane);
                v16bf bB = load_b_frag(lB, nGrp * 32 + 16, kk, lane);
                t[0][0] = wmma_bf16(a0, bA, t[0][0]);
                t[0][1] = wmma_bf16(a0, bB, t[0][1]);
                t[1][0] = wmma_bf16(a1, bA, t[1][0]);
                t[1][1] = wmma_bf16(a1, bB, t[1][1]);
            }
            __syncthreads();
        }
#pragma unroll
        for (int mi = 0; mi < 2; ++mi)
#pragma unroll
            for (int v = 0; v < 8; ++v) {
                int r = mGrp * 32 + mi * 16 + v + rowOff;   // row within tile
                float sc = lS[r * NEXP + e];                // sign already folded in
                o[mi][0][v] += sc * t[mi][0][v];
                o[mi][1][v] += sc * t[mi][1][v];
            }
    }

    float ss = 0.0f;
#pragma unroll
    for (int mi = 0; mi < 2; ++mi)
#pragma unroll
        for (int v = 0; v < 8; ++v) {
            int rg = m0 + mGrp * 32 + mi * 16 + v + rowOff;
            size_t base = (size_t)rg * DDIM + n0 + nGrp * 32;
            out[base + col]      = o[mi][0][v];
            out[base + 16 + col] = o[mi][1][v];
            ss += o[mi][0][v] * o[mi][0][v] + o[mi][1][v] * o[mi][1][v];
        }
#pragma unroll
    for (int off = 16; off > 0; off >>= 1) ss += __shfl_down(ss, off, 32);
    if (lane == 0) wred[wave] = ss;
    __syncthreads();
    if (tid == 0) {
        float t = 0.0f;
        for (int w = 0; w < 8; ++w) t += wred[w];
        atomicAdd(sumsq, t);
    }
}

__global__ void fin_kernel(const float* sumsq, float* out_tail) {
    *out_tail = *sumsq * (1.0f / ((float)MTOK * (float)DDIM));
}

// ---------------- launch ----------------
extern "C" void kernel_launch(void* const* d_in, const int* in_sizes, int n_in,
                              void* d_out, int out_size, void* d_ws, size_t ws_size,
                              hipStream_t stream) {
    const float* x  = (const float*)d_in[0];
    const float* Wr = (const float*)d_in[1];
    const float* Wg = (const float*)d_in[2];
    const float* Wu = (const float*)d_in[3];
    const float* Wd = (const float*)d_in[4];
    float* out = (float*)d_out;

    // workspace layout (~256.3 MB): sumsq | wtok | xb | wgT | wuT | wdT | hb
    char* ws = (char*)d_ws;
    float* sumsq = (float*)ws;
    float* wtok  = (float*)(ws + 256);
    unsigned short* xb  = (unsigned short*)(ws + 256 + (size_t)MTOK * NEXP * 4);
    unsigned short* wgT = xb  + (size_t)MTOK * DDIM;
    unsigned short* wuT = wgT + (size_t)NEXP * DDIM * IDIM;
    unsigned short* wdT = wuT + (size_t)NEXP * DDIM * IDIM;
    unsigned short* hb  = wdT + (size_t)NEXP * IDIM * DDIM;

    zero_kernel<<<1, 1, 0, stream>>>(sumsq);

    const int n8x = MTOK * DDIM / 8;
    cvt_kernel<<<(n8x + 255) / 256, 256, 0, stream>>>(x, xb, n8x);
    // Wg,Wu: (E, K=D, N=I) -> (E, I, D);  Wd: (E, K=I, N=D) -> (E, D, I)
    cvtT_kernel<<<dim3(IDIM / 32, DDIM / 32, NEXP), 256, 0, stream>>>(Wg, wgT, DDIM, IDIM);
    cvtT_kernel<<<dim3(IDIM / 32, DDIM / 32, NEXP), 256, 0, stream>>>(Wu, wuT, DDIM, IDIM);
    cvtT_kernel<<<dim3(DDIM / 32, IDIM / 32, NEXP), 256, 0, stream>>>(Wd, wdT, IDIM, DDIM);

    router_kernel<<<MTOK, 256, 0, stream>>>(x, Wr, wtok);

    gemm1_kernel<<<dim3(IDIM / 64, MTOK / 128, NEXP), 256, 0, stream>>>(xb, wgT, wuT, hb);
    gemm2_kernel<<<dim3(DDIM / 64, MTOK / 128), 256, 0, stream>>>(hb, wdT, wtok, out, sumsq);

    fin_kernel<<<1, 1, 0, stream>>>(sumsq, out + (size_t)MTOK * DDIM);
}
